// SetAbstractionLayer_63702954934284
// MI455X (gfx1250) — matrix operand order repaired and checked by reading, not compiled, not run on hardware
//
#include <hip/hip_runtime.h>
#include <hip/hip_bf16.h>
#include <math.h>
#include <stdint.h>

typedef _Float16 fp16;
typedef __attribute__((ext_vector_type(16))) _Float16 v16h;
typedef __attribute__((ext_vector_type(8)))  _Float16 v8h;
typedef __attribute__((ext_vector_type(8)))  float    v8f;

#define B_   16
#define N_   4096
#define S_   1024
#define KS_  32
#define CF_  64
#define M_   (B_ * S_ * KS_)   /* 524288 rows through the MLP */
#define RAD2 0.25f

static __device__ __forceinline__ float gelu_exact(float x) {
    return 0.5f * x * (1.0f + erff(x * 0.70710678118654752f));
}

// ---------------------------------------------------------------------------
// 1) Farthest point sampling: one block per batch, xyz cached in LDS (48 KB).
//    1024 serial iterations; block argmax via wave32 shuffles + LDS tree.
//    Tie-break = smallest index (matches jnp.argmax).
// ---------------------------------------------------------------------------
__global__ __launch_bounds__(1024) void fps_kernel(const float* __restrict__ xyz,
                                                   int* __restrict__ fps_idx) {
    __shared__ float sx[N_], sy[N_], sz[N_];
    __shared__ float rd[32];
    __shared__ int   ri[32];
    __shared__ int   s_far;
    const int b   = blockIdx.x;
    const int tid = threadIdx.x;
    const float* bx = xyz + (size_t)b * N_ * 3;
    for (int p = tid; p < N_; p += 1024) {
        sx[p] = bx[p * 3 + 0];
        sy[p] = bx[p * 3 + 1];
        sz[p] = bx[p * 3 + 2];
    }
    if (tid == 0) s_far = 0;
    __syncthreads();

    float px[4], py[4], pz[4], dist[4];
#pragma unroll
    for (int j = 0; j < 4; ++j) {
        int p = tid + j * 1024;
        px[j] = sx[p]; py[j] = sy[p]; pz[j] = sz[p];
        dist[j] = 1e10f;
    }
    const int lane = tid & 31, wid = tid >> 5;

    for (int it = 0; it < S_; ++it) {
        int far = s_far;
        if (tid == 0) fps_idx[b * S_ + it] = far;
        float cx = sx[far], cy = sy[far], cz = sz[far];
        float bd = -1.0f; int bi = 0;
#pragma unroll
        for (int j = 0; j < 4; ++j) {
            float dx = px[j] - cx, dy = py[j] - cy, dz = pz[j] - cz;
            float d = dx * dx + dy * dy + dz * dz;
            dist[j] = fminf(dist[j], d);
            if (dist[j] > bd) { bd = dist[j]; bi = tid + j * 1024; }
        }
        // wave32 reduce: larger distance wins, ties -> smaller index
#pragma unroll
        for (int off = 16; off > 0; off >>= 1) {
            float od = __shfl_down(bd, off, 32);
            int   oi = __shfl_down(bi, off, 32);
            if (od > bd || (od == bd && oi < bi)) { bd = od; bi = oi; }
        }
        if (lane == 0) { rd[wid] = bd; ri[wid] = bi; }
        __syncthreads();
        if (wid == 0) {
            float d2 = rd[lane]; int i2 = ri[lane];
#pragma unroll
            for (int off = 16; off > 0; off >>= 1) {
                float od = __shfl_down(d2, off, 32);
                int   oi = __shfl_down(i2, off, 32);
                if (od > d2 || (od == d2 && oi < i2)) { d2 = od; i2 = oi; }
            }
            if (lane == 0) s_far = i2;
        }
        __syncthreads();
    }
}

// ---------------------------------------------------------------------------
// 2) new_xyz = xyz[fps_idx]   (also the query points for ball query)
// ---------------------------------------------------------------------------
__global__ void newxyz_kernel(const float* __restrict__ xyz,
                              const int* __restrict__ fps_idx,
                              float* __restrict__ out) {
    int i = blockIdx.x * blockDim.x + threadIdx.x;
    if (i >= B_ * S_ * 3) return;
    int bs = i / 3, c = i % 3;
    int b  = bs >> 10;
    int qi = fps_idx[bs];
    out[i] = xyz[((size_t)b * N_ + qi) * 3 + c];
}

// ---------------------------------------------------------------------------
// 3) Ball query: one wave32 per query. Ballot + prefix popcount gives the
//    first-32 in-radius indices in ascending order; fill remainder with first.
// ---------------------------------------------------------------------------
__global__ __launch_bounds__(128) void ballquery_kernel(const float* __restrict__ xyz,
                                                        const int* __restrict__ fps_idx,
                                                        int* __restrict__ gidx) {
    const int lane = threadIdx.x & 31;
    const int gw   = blockIdx.x * 4 + (threadIdx.x >> 5);
    if (gw >= B_ * S_) return;
    const int b = gw >> 10;
    const float* bx = xyz + (size_t)b * N_ * 3;
    const int qi = fps_idx[gw];
    const float qx = bx[qi * 3 + 0], qy = bx[qi * 3 + 1], qz = bx[qi * 3 + 2];
    int cnt = 0, first = -1;
    int* out = gidx + (size_t)gw * KS_;
    for (int c0 = 0; c0 < N_; c0 += 32) {
        int p = c0 + lane;
        float dx = bx[p * 3 + 0] - qx;
        float dy = bx[p * 3 + 1] - qy;
        float dz = bx[p * 3 + 2] - qz;
        float d  = dx * dx + dy * dy + dz * dz;
        bool inr = d <= RAD2;
        unsigned mask = (unsigned)__ballot(inr);
        if (first < 0 && mask) first = c0 + (__ffs(mask) - 1);
        if (inr) {
            int rank = cnt + __popc(mask & ((1u << lane) - 1u));
            if (rank < KS_) out[rank] = p;
        }
        cnt += __popc(mask);
        if (cnt >= KS_) break;
    }
    if (lane >= cnt) out[lane] = first;  // cnt >= 1 always (query point itself)
}

// ---------------------------------------------------------------------------
// 4) Gather + concat -> layer-0 activation, f16, K padded 67 -> 96
// ---------------------------------------------------------------------------
__global__ void gather_kernel(const float* __restrict__ xyz,
                              const float* __restrict__ feat,
                              const int* __restrict__ gidx,
                              const float* __restrict__ newxyz,
                              fp16* __restrict__ A) {
    long long i = (long long)blockIdx.x * blockDim.x + threadIdx.x;
    if (i >= (long long)M_ * 96) return;
    int row = (int)(i / 96);
    int c   = (int)(i % 96);
    int b   = row >> 15;          // / (S_*KS_)
    int rs  = row & 32767;
    int s   = rs >> 5;
    int g   = gidx[row];
    float v;
    if (c < 3) {
        v = xyz[((size_t)b * N_ + g) * 3 + c] - newxyz[((size_t)b * S_ + s) * 3 + c];
    } else if (c < 67) {
        v = feat[((size_t)b * N_ + g) * CF_ + (c - 3)];
    } else {
        v = 0.0f;
    }
    A[(size_t)row * 96 + c] = (fp16)v;
}

// ---------------------------------------------------------------------------
// 5) Weight prep: w (cout x cin) f32 -> f16, zero-padded to Kpad
// ---------------------------------------------------------------------------
__global__ void wprep_kernel(const float* __restrict__ w, fp16* __restrict__ Wp,
                             int Cin, int Kpad, int Nout) {
    int i = blockIdx.x * blockDim.x + threadIdx.x;
    if (i >= Nout * Kpad) return;
    int n = i / Kpad, k = i % Kpad;
    Wp[i] = (k < Cin) ? (fp16)w[n * Cin + k] : (fp16)0.0f;
}

// ---------------------------------------------------------------------------
// 6) GEMM via V_WMMA_F32_16X16X32_F16. One wave per 16-row tile; A fragments
//    loaded once and reused across all N-tiles. H = A @ W^T + bias (f16 out).
//    Fragment layouts per CDNA5 ISA 7.12.2:
//      A 16x32 f16 : lane(m=lane&15, khalf=lane>>4), k(e)=e+8*khalf+(e>=8?8:0)
//      B 32x16 f16 : lane(n=lane&15, khalf),          k(e)=e+8*khalf+(e>=8?8:0)
//      C/D 16x16   : m = r + 8*khalf, n = lane&15
// ---------------------------------------------------------------------------
template <int KT, int NT>
__global__ __launch_bounds__(128) void gemm_wmma(const fp16* __restrict__ A,
                                                 const fp16* __restrict__ W,
                                                 const float* __restrict__ bias,
                                                 fp16* __restrict__ H) {
    constexpr int Kpad = KT * 32;
    constexpr int Nout = NT * 16;
    const int lane  = threadIdx.x & 31;
    const int mt    = blockIdx.x * 4 + (threadIdx.x >> 5);
    const int mrow  = lane & 15;
    const int khalf = lane >> 4;

    const fp16* arow = A + (size_t)(mt * 16 + mrow) * Kpad;
    v16h afrag[KT];
#pragma unroll
    for (int kt = 0; kt < KT; ++kt) {
        union { v16h v; v8h h[2]; } u;
        u.h[0] = *(const v8h*)(arow + kt * 32 + khalf * 8);
        u.h[1] = *(const v8h*)(arow + kt * 32 + khalf * 8 + 16);
        afrag[kt] = u.v;
    }

#pragma unroll
    for (int nt = 0; nt < NT; ++nt) {
        const fp16* wrow = W + (size_t)(nt * 16 + mrow) * Kpad;  // row n of w == column n of B
        v8f acc = {};
#pragma unroll
        for (int kt = 0; kt < KT; ++kt) {
            union { v16h v; v8h h[2]; } ub;
            ub.h[0] = *(const v8h*)(wrow + kt * 32 + khalf * 8);
            ub.h[1] = *(const v8h*)(wrow + kt * 32 + khalf * 8 + 16);
            v16h bfr = ub.v;
            acc = __builtin_amdgcn_wmma_f32_16x16x32_f16(
                false, afrag[kt], false, bfr, (short)0, acc, false, false);
        }
        const int   ng = nt * 16 + mrow;
        const float bn = bias[ng];
        fp16* hp = H + (size_t)(mt * 16 + khalf * 8) * Nout + ng;
#pragma unroll
        for (int r = 0; r < 8; ++r)
            hp[(size_t)r * Nout] = (fp16)(acc[r] + bn);
    }
}

// ---------------------------------------------------------------------------
// 7) Batch-norm statistics: per-channel sum / sumsq via LDS atomics + one
//    global atomic per channel per block.
// ---------------------------------------------------------------------------
__global__ __launch_bounds__(256) void zero_stats_kernel(float* __restrict__ s) {
    s[threadIdx.x] = 0.0f;  // 256 floats = sum[128] ++ sumsq[128]
}

__global__ __launch_bounds__(256) void stats_kernel(const fp16* __restrict__ H,
                                                    float* __restrict__ gsum,
                                                    float* __restrict__ gsumsq,
                                                    int Nout, long long total) {
    __shared__ float ssum[128], ssq[128];
    for (int c = threadIdx.x; c < Nout; c += 256) { ssum[c] = 0.f; ssq[c] = 0.f; }
    __syncthreads();
    long long stride = (long long)gridDim.x * 256;
    for (long long i = (long long)blockIdx.x * 256 + threadIdx.x; i < total; i += stride) {
        float v = (float)H[i];
        int c = (int)(i % Nout);
        atomicAdd(&ssum[c], v);
        atomicAdd(&ssq[c], v * v);
    }
    __syncthreads();
    for (int c = threadIdx.x; c < Nout; c += 256) {
        atomicAdd(&gsum[c], ssum[c]);
        atomicAdd(&gsumsq[c], ssq[c]);
    }
}

__global__ void finalize_stats_kernel(const float* __restrict__ gsum,
                                      const float* __restrict__ gsumsq,
                                      const float* __restrict__ gamma,
                                      const float* __restrict__ beta,
                                      float* __restrict__ scale,
                                      float* __restrict__ shift, int Nout) {
    int c = blockIdx.x * blockDim.x + threadIdx.x;
    if (c >= Nout) return;
    float inv  = 1.0f / (float)M_;
    float mean = gsum[c] * inv;
    float var  = fmaxf(gsumsq[c] * inv - mean * mean, 0.0f);
    float sc   = gamma[c] * rsqrtf(var + 1e-5f);
    scale[c] = sc;
    shift[c] = beta[c] - mean * sc;
}

// ---------------------------------------------------------------------------
// 8) Fused normalize + exact GELU -> next layer activation (f16)
// ---------------------------------------------------------------------------
__global__ void norm_gelu_kernel(const fp16* __restrict__ H,
                                 const float* __restrict__ scale,
                                 const float* __restrict__ shift,
                                 fp16* __restrict__ Aout, int Nout, long long total) {
    long long i = (long long)blockIdx.x * blockDim.x + threadIdx.x;
    if (i >= total) return;
    int c = (int)(i % Nout);
    float x = (float)H[i] * scale[c] + shift[c];
    Aout[i] = (fp16)gelu_exact(x);
}

// ---------------------------------------------------------------------------
// 9) Final layer: fused normalize + GELU + max over the 32 group samples
// ---------------------------------------------------------------------------
__global__ void maxpool_kernel(const fp16* __restrict__ H,
                               const float* __restrict__ scale,
                               const float* __restrict__ shift,
                               float* __restrict__ out) {
    int i = blockIdx.x * blockDim.x + threadIdx.x;
    if (i >= B_ * S_ * 128) return;
    int d  = i & 127;
    int bs = i >> 7;
    float sc = scale[d], sh = shift[d];
    float m = -3.4e38f;
    const fp16* hp = H + (size_t)bs * KS_ * 128 + d;
#pragma unroll 4
    for (int k = 0; k < KS_; ++k) {
        float x = (float)hp[(size_t)k * 128] * sc + sh;
        m = fmaxf(m, gelu_exact(x));
    }
    out[i] = m;
}

// ---------------------------------------------------------------------------
// Workspace layout (bytes):
//   fps_idx   @ 0          (64 KB)
//   group_idx @ 65536      (2 MB)
//   sum/sumsq @ 2162688    (1 KB)
//   scale     @ 2163712, shift @ 2164224
//   Wpad f16  @ 2164736    (24 KB)
//   A   f16   @ 2189312    (M*96*2  =  96 MB)
//   H   f16   @ 102852608  (M*128*2 = 128 MB)
// total ~226 MB
// ---------------------------------------------------------------------------
extern "C" void kernel_launch(void* const* d_in, const int* in_sizes, int n_in,
                              void* d_out, int out_size, void* d_ws, size_t ws_size,
                              hipStream_t stream) {
    const float* xyz  = (const float*)d_in[0];
    const float* feat = (const float*)d_in[1];
    const float* w[3]  = {(const float*)d_in[2],  (const float*)d_in[6],  (const float*)d_in[10]};
    const float* bb[3] = {(const float*)d_in[3],  (const float*)d_in[7],  (const float*)d_in[11]};
    const float* gg[3] = {(const float*)d_in[4],  (const float*)d_in[8],  (const float*)d_in[12]};
    const float* be[3] = {(const float*)d_in[5],  (const float*)d_in[9],  (const float*)d_in[13]};

    uint8_t* ws = (uint8_t*)d_ws;
    int*   fps   = (int*)(ws + 0);
    int*   gidx  = (int*)(ws + 65536);
    float* gsum  = (float*)(ws + 2162688);
    float* gsq   = gsum + 128;
    float* scale = (float*)(ws + 2163712);
    float* shift = (float*)(ws + 2164224);
    fp16*  Wp    = (fp16*)(ws + 2164736);
    fp16*  Abuf  = (fp16*)(ws + 2189312);
    fp16*  Hbuf  = (fp16*)(ws + 102852608ULL);

    float* out_xyz  = (float*)d_out;
    float* out_feat = out_xyz + B_ * S_ * 3;

    fps_kernel<<<B_, 1024, 0, stream>>>(xyz, fps);
    newxyz_kernel<<<(B_ * S_ * 3 + 255) / 256, 256, 0, stream>>>(xyz, fps, out_xyz);
    ballquery_kernel<<<(B_ * S_) / 4, 128, 0, stream>>>(xyz, fps, gidx);
    {
        long long tot = (long long)M_ * 96;
        gather_kernel<<<(unsigned)((tot + 255) / 256), 256, 0, stream>>>(xyz, feat, gidx, out_xyz, Abuf);
    }

    const int Cin[3] = {67, 64, 64};
    const int Kp[3]  = {96, 64, 64};
    const int No[3]  = {64, 64, 128};
    for (int l = 0; l < 3; ++l) {
        wprep_kernel<<<(No[l] * Kp[l] + 255) / 256, 256, 0, stream>>>(w[l], Wp, Cin[l], Kp[l], No[l]);
        if (l == 0)      gemm_wmma<3, 4><<<M_ / 64, 128, 0, stream>>>(Abuf, Wp, bb[l], Hbuf);
        else if (l == 1) gemm_wmma<2, 4><<<M_ / 64, 128, 0, stream>>>(Abuf, Wp, bb[l], Hbuf);
        else             gemm_wmma<2, 8><<<M_ / 64, 128, 0, stream>>>(Abuf, Wp, bb[l], Hbuf);

        zero_stats_kernel<<<1, 256, 0, stream>>>(gsum);
        long long tot = (long long)M_ * No[l];
        stats_kernel<<<4096, 256, 0, stream>>>(Hbuf, gsum, gsq, No[l], tot);
        finalize_stats_kernel<<<1, 128, 0, stream>>>(gsum, gsq, gg[l], be[l], scale, shift, No[l]);

        if (l < 2) {
            norm_gelu_kernel<<<(unsigned)((tot + 255) / 256), 256, 0, stream>>>(
                Hbuf, scale, shift, Abuf, No[l], tot);
        } else {
            maxpool_kernel<<<(B_ * S_ * 128 + 255) / 256, 256, 0, stream>>>(
                Hbuf, scale, shift, out_feat);
        }
    }
}
